// ML_LSTM_S2S_10814727652037
// MI455X (gfx1250) — compile-verified, matrix operand order checked by reading, not run on hardware
//
#include <hip/hip_runtime.h>
#include <hip/hip_bf16.h>

#define B_ 256
#define T_ 512
#define F_ 64
#define U_ 256
#define S_ 64
#define O_ 8

typedef __attribute__((ext_vector_type(16))) __bf16 bf16x16;
typedef __attribute__((ext_vector_type(8)))  float  f32x8;
typedef __attribute__((ext_vector_type(4))) unsigned int uint32x4;
typedef __attribute__((ext_vector_type(8))) int int32x8;
typedef __attribute__((ext_vector_type(4))) int int32x4;

#if __has_builtin(__builtin_amdgcn_tensor_load_to_lds)
#define USE_TDM 1
#else
#define USE_TDM 0
#endif

union FragBF { bf16x16 v; uint4 q[2]; };

__device__ __forceinline__ bf16x16 frag_zero() {
  FragBF f;
  f.q[0] = make_uint4(0u, 0u, 0u, 0u);
  f.q[1] = make_uint4(0u, 0u, 0u, 0u);
  return f.v;
}

// Load a 16x32 bf16 fragment (A layout, or B layout from an [N,K] row-major
// weight). Lane l owns row (l&15); half=(l>>4) selects K chunks
// [8h,8h+8) and [16+8h,16+8h+8)  (ISA 7.12.2, 16-bit A-matrix 16x32).
__device__ __forceinline__ bf16x16 frag_load(const __hip_bfloat16* base, size_t ld, int lane) {
  const int r  = lane & 15;
  const int hh = (lane >> 4) & 1;
  const __hip_bfloat16* p = base + (size_t)r * ld + 8 * hh;
  FragBF f;
  f.q[0] = *reinterpret_cast<const uint4*>(p);
  f.q[1] = *reinterpret_cast<const uint4*>(p + 16);
  return f.v;
}

__device__ __forceinline__ float sigm_(float x) { return 1.0f / (1.0f + __expf(-x)); }
__device__ __forceinline__ float tanh_(float x) { return 2.0f / (1.0f + __expf(-2.0f * x)) - 1.0f; }

#if USE_TDM
// TDM 2D tile load: rows x cols (bf16, 2B elements), packed row-major into LDS.
// Descriptor layout per CDNA5 ISA (D# groups 0/1).
__device__ __forceinline__ void tdm_load_2d(unsigned int lds_byte_off, const void* gptr,
                                            unsigned int rows, unsigned int cols,
                                            unsigned long long row_stride_elts) {
  unsigned long long ga = (unsigned long long)(uintptr_t)gptr;
  uint32x4 g0;
  g0[0] = 1u;                                            // count=1, user-mode D#
  g0[1] = lds_byte_off;                                  // lds_addr (bytes)
  g0[2] = (unsigned int)(ga & 0xFFFFFFFFu);              // global_addr[31:0]
  g0[3] = (unsigned int)((ga >> 32) & 0x01FFFFFFu)       // global_addr[56:32]
          | (2u << 30);                                  // type = 2 ("image")
  int32x8 g1;
  g1[0] = (int)(1u << 16);                               // data_size=1 (2 bytes), wg_mask=0
  g1[1] = (int)((cols & 0xFFFFu) << 16);                 // tensor_dim0[15:0]
  g1[2] = (int)(((cols >> 16) & 0xFFFFu)                 // tensor_dim0[31:16]
          | ((rows & 0xFFFFu) << 16));                   // tensor_dim1[15:0]
  g1[3] = (int)(((rows >> 16) & 0xFFFFu)                 // tensor_dim1[31:16]
          | ((cols & 0xFFFFu) << 16));                   // tile_dim0
  g1[4] = (int)(rows & 0xFFFFu);                         // tile_dim1 (tile_dim2=0)
  g1[5] = (int)(unsigned int)(row_stride_elts & 0xFFFFFFFFu);   // dim0_stride[31:0]
  g1[6] = (int)(unsigned int)((row_stride_elts >> 32) & 0xFFFFu); // dim0_stride[47:32]
  g1[7] = 0;
  int32x4 z4 = {0, 0, 0, 0};
#if defined(__clang_major__) && (__clang_major__ >= 23)
  int32x8 z8 = {0, 0, 0, 0, 0, 0, 0, 0};
  __builtin_amdgcn_tensor_load_to_lds(g0, g1, z4, z4, z8, 0);
#else
  __builtin_amdgcn_tensor_load_to_lds(g0, g1, z4, z4, 0);
#endif
}
#endif

// ---------------- prep kernels ----------------

__global__ void cvt_f32_bf16(const float* __restrict__ in, __hip_bfloat16* __restrict__ out, int n) {
  for (int i = blockIdx.x * blockDim.x + threadIdx.x; i < n; i += gridDim.x * blockDim.x)
    out[i] = __float2bfloat16(in[i]);
}

// cnn_w [5][64][256] -> wT [256][320]  (row = out unit, K = tap*64+f)
__global__ void prep_conv_w(const float* __restrict__ w, __hip_bfloat16* __restrict__ wT) {
  int idx = blockIdx.x * blockDim.x + threadIdx.x;
  if (idx >= 256 * 320) return;
  int u  = idx / 320;
  int kk = idx - u * 320;
  int tap = kk >> 6;
  int f   = kk & 63;
  wT[idx] = __float2bfloat16(w[((size_t)tap * 64 + f) * 256 + u]);
}

// enc_w [256][1024], enc_u [256][1024] -> wT [1024][512] (row n, K: 0..255 = W, 256..511 = U)
__global__ void prep_enc_w(const float* __restrict__ w, const float* __restrict__ u,
                           __hip_bfloat16* __restrict__ wT) {
  int idx = blockIdx.x * blockDim.x + threadIdx.x;
  if (idx >= 1024 * 512) return;
  int n = idx >> 9;
  int k = idx & 511;
  float v = (k < 256) ? w[(size_t)k * 1024 + n] : u[(size_t)(k - 256) * 1024 + n];
  wT[idx] = __float2bfloat16(v);
}

// decoder: z = h@(cell_w + cell_u) + b  -> wT [1024][256]
__global__ void prep_dec_w(const float* __restrict__ w, const float* __restrict__ u,
                           __hip_bfloat16* __restrict__ wT) {
  int idx = blockIdx.x * blockDim.x + threadIdx.x;
  if (idx >= 1024 * 256) return;
  int n = idx >> 8;
  int k = idx & 255;
  wT[idx] = __float2bfloat16(w[(size_t)k * 1024 + n] + u[(size_t)k * 1024 + n]);
}

// ---------------- conv1d (same pad, k=5) + ReLU via WMMA ----------------
__global__ __launch_bounds__(256) void conv_relu_wmma(
    const __hip_bfloat16* __restrict__ xb,   // [B,T,64]
    const __hip_bfloat16* __restrict__ wT,   // [256][320]
    const float* __restrict__ bias,          // [256]
    __hip_bfloat16* __restrict__ y)          // [B,T,256]
{
  const int wave = threadIdx.x >> 5;
  const int lane = threadIdx.x & 31;
  const int tile = blockIdx.x * (blockDim.x >> 5) + wave;
  const int ntile = tile & 15;
  const int mtile = tile >> 4;
  const int r  = lane & 15;
  const int hh = (lane >> 4) & 1;
  const int bt = mtile * 16 + r;
  const int b  = bt >> 9;
  const int t  = bt & 511;

  f32x8 acc = {};
#pragma unroll
  for (int kt = 0; kt < 10; ++kt) {          // K = 320 = 10 tiles of 32
    const int tap = kt >> 1;
    const int tt = t + tap - 2;
    bf16x16 a;
    if (tt >= 0 && tt < T_) {
      const int f0 = ((kt & 1) << 5) + 8 * hh;
      const __hip_bfloat16* p = xb + ((size_t)b * T_ + tt) * F_ + f0;
      FragBF fa;
      fa.q[0] = *reinterpret_cast<const uint4*>(p);
      fa.q[1] = *reinterpret_cast<const uint4*>(p + 16);
      a = fa.v;
    } else {
      a = frag_zero();
    }
    bf16x16 bw = frag_load(wT + (size_t)(ntile * 16) * 320 + kt * 32, 320, lane);
    acc = __builtin_amdgcn_wmma_f32_16x16x32_bf16(false, a, false, bw, (short)0, acc, false, false);
  }

  const int col = ntile * 16 + (lane & 15);
  const float bv = bias[col];
  const size_t row0 = (size_t)mtile * 16 + 8 * hh;
#pragma unroll
  for (int v = 0; v < 8; ++v) {
    float val = fmaxf(acc[v] + bv, 0.0f);
    y[(row0 + v) * U_ + col] = __float2bfloat16(val);
  }
}

// ---------------- fused LSTM step via WMMA + TDM-staged A operand ----------------
// Block b covers batch-row tile m = b>>1 and 8 unit-column tiles. The shared
// [x_t | h] A tile (16 x 512 bf16) is DMA'd into LDS once per block by the
// Tensor Data Mover, then all 8 waves read ds_load_b128 fragments from it.
__global__ __launch_bounds__(256) void lstm_step_wmma(
    const __hip_bfloat16* __restrict__ x,    // [B] rows (stride x_stride) or nullptr
    size_t x_stride,
    const __hip_bfloat16* __restrict__ hin,  // [B,256] bf16
    const float* __restrict__ cin,           // [B,256] f32
    const __hip_bfloat16* __restrict__ wT,   // [1024][Ktot]
    const float* __restrict__ bias,          // [1024] (i,f,g,o)
    __hip_bfloat16* __restrict__ hout,       // [B,256] bf16
    float* __restrict__ cout,                // [B,256] f32
    __hip_bfloat16* __restrict__ yout,       // seq out (stride y_stride) or nullptr
    size_t y_stride,
    float* __restrict__ fout)                // f32 h out (decoder) or nullptr
{
  __shared__ __hip_bfloat16 Ax[16 * U_];     // x_t tile (encoder only)
  __shared__ __hip_bfloat16 Ah[16 * U_];     // h tile

  const int wave = threadIdx.x >> 5;
  const int lane = threadIdx.x & 31;
  const int tile = blockIdx.x * (blockDim.x >> 5) + wave;   // 256 tiles
  const int j = tile & 15;                   // unit col tile
  const int m = tile >> 4;                   // batch row tile (uniform per block)
  const bool has_x = (x != nullptr);
  const int ktiles = has_x ? 16 : 8;
  const size_t ldw = (size_t)ktiles * 32;

#if USE_TDM
  if (threadIdx.x == 0) {
    if (has_x)
      tdm_load_2d((unsigned int)(uintptr_t)&Ax[0], x + (size_t)(m * 16) * x_stride,
                  16u, (unsigned int)U_, (unsigned long long)x_stride);
    tdm_load_2d((unsigned int)(uintptr_t)&Ah[0], hin + (size_t)(m * 16) * U_,
                16u, (unsigned int)U_, (unsigned long long)U_);
    __builtin_amdgcn_s_wait_tensorcnt(0);
  }
#else
  {
    const int nvec = 16 * U_ / 8;            // 16B chunks per tile
    for (int i = threadIdx.x; i < nvec; i += blockDim.x) {
      const int rr = i >> 5, cc = (i & 31) * 8;
      if (has_x)
        *reinterpret_cast<uint4*>(&Ax[rr * U_ + cc]) =
            *reinterpret_cast<const uint4*>(x + (size_t)(m * 16 + rr) * x_stride + cc);
      *reinterpret_cast<uint4*>(&Ah[rr * U_ + cc]) =
          *reinterpret_cast<const uint4*>(hin + (size_t)(m * 16 + rr) * U_ + cc);
    }
  }
#endif
  __syncthreads();

  f32x8 aI = {}, aF = {}, aG = {}, aO = {};
  const __hip_bfloat16* wbase = wT + (size_t)(j * 16) * ldw;

  for (int kt = 0; kt < ktiles; ++kt) {
    bf16x16 a;
    if (has_x && kt < 8) {
      a = frag_load(Ax + kt * 32, U_, lane);
    } else {
      const int kh = has_x ? (kt - 8) : kt;
      a = frag_load(Ah + kh * 32, U_, lane);
    }
    const __hip_bfloat16* wrow = wbase + kt * 32;
    if (kt + 1 < ktiles)
      __builtin_prefetch(wrow + 32, 0, 3);   // global_prefetch_b8 on gfx1250
    bf16x16 bI = frag_load(wrow,                     ldw, lane);
    bf16x16 bF = frag_load(wrow + (size_t)256 * ldw, ldw, lane);
    bf16x16 bG = frag_load(wrow + (size_t)512 * ldw, ldw, lane);
    bf16x16 bO = frag_load(wrow + (size_t)768 * ldw, ldw, lane);
    aI = __builtin_amdgcn_wmma_f32_16x16x32_bf16(false, a, false, bI, (short)0, aI, false, false);
    aF = __builtin_amdgcn_wmma_f32_16x16x32_bf16(false, a, false, bF, (short)0, aF, false, false);
    aG = __builtin_amdgcn_wmma_f32_16x16x32_bf16(false, a, false, bG, (short)0, aG, false, false);
    aO = __builtin_amdgcn_wmma_f32_16x16x32_bf16(false, a, false, bO, (short)0, aO, false, false);
  }

  const int col = j * 16 + (lane & 15);
  const int hh  = (lane >> 4) & 1;
  const float bI_ = bias[col];
  const float bF_ = bias[256 + col];
  const float bG_ = bias[512 + col];
  const float bO_ = bias[768 + col];

#pragma unroll
  for (int v = 0; v < 8; ++v) {
    const size_t row = (size_t)m * 16 + v + 8 * hh;
    const float cprev = cin[row * U_ + col];
    const float iv = sigm_(aI[v] + bI_);
    const float fv = sigm_(aF[v] + bF_);
    const float gv = tanh_(aG[v] + bG_);
    const float ov = sigm_(aO[v] + bO_);
    const float cn = fv * cprev + iv * gv;
    const float hn = ov * tanh_(cn);
    cout[row * U_ + col] = cn;
    const __hip_bfloat16 hb = __float2bfloat16(hn);
    hout[row * U_ + col] = hb;
    if (yout) yout[row * y_stride + col] = hb;
    if (fout) fout[row * U_ + col] = hn;
  }
}

// ---------------- dense head: [64,B,256] f32 @ [256,8] + b -> [B,64,8] ----------------
__global__ void dense_out_kernel(const float* __restrict__ hseq,
                                 const float* __restrict__ w,
                                 const float* __restrict__ bias,
                                 float* __restrict__ out) {
  int idx = blockIdx.x * blockDim.x + threadIdx.x;   // B*64*8
  if (idx >= B_ * S_ * O_) return;
  const int o = idx & 7;
  const int s = (idx >> 3) & 63;
  const int b = idx >> 9;
  const float* hp = hseq + ((size_t)s * B_ + b) * U_;
  float acc = bias[o];
#pragma unroll 8
  for (int u = 0; u < U_; ++u) acc = fmaf(hp[u], w[(size_t)u * O_ + o], acc);
  out[idx] = acc;
}

// ---------------- host ----------------

extern "C" void kernel_launch(void* const* d_in, const int* in_sizes, int n_in,
                              void* d_out, int out_size, void* d_ws, size_t ws_size,
                              hipStream_t stream) {
  (void)in_sizes; (void)n_in; (void)out_size; (void)ws_size;

  const float* x     = (const float*)d_in[0];
  const float* cnn_w = (const float*)d_in[1];
  const float* cnn_b = (const float*)d_in[2];
  const float* ew[3] = {(const float*)d_in[3], (const float*)d_in[6], (const float*)d_in[9]};
  const float* eu[3] = {(const float*)d_in[4], (const float*)d_in[7], (const float*)d_in[10]};
  const float* eb[3] = {(const float*)d_in[5], (const float*)d_in[8], (const float*)d_in[11]};
  const float* cw    = (const float*)d_in[12];
  const float* cu    = (const float*)d_in[13];
  const float* cb    = (const float*)d_in[14];
  const float* dw    = (const float*)d_in[15];
  const float* db    = (const float*)d_in[16];
  float* out = (float*)d_out;

  // workspace carve (256B aligned)
  char* wp = (char*)d_ws;
  auto carve = [&](size_t bytes) { char* p = wp; wp += (bytes + 255) & ~(size_t)255; return p; };
  __hip_bfloat16* xb   = (__hip_bfloat16*)carve((size_t)B_ * T_ * F_ * 2);   // 16 MB
  __hip_bfloat16* y0   = (__hip_bfloat16*)carve((size_t)B_ * T_ * U_ * 2);   // 64 MB
  __hip_bfloat16* y1   = (__hip_bfloat16*)carve((size_t)B_ * T_ * U_ * 2);   // 64 MB
  __hip_bfloat16* hpp[2] = {(__hip_bfloat16*)carve((size_t)B_ * U_ * 2),
                            (__hip_bfloat16*)carve((size_t)B_ * U_ * 2)};
  float* cpp[2] = {(float*)carve((size_t)B_ * U_ * 4), (float*)carve((size_t)B_ * U_ * 4)};
  float* dec    = (float*)carve((size_t)S_ * B_ * U_ * 4);                   // 16 MB
  __hip_bfloat16* cwT = (__hip_bfloat16*)carve((size_t)256 * 320 * 2);
  __hip_bfloat16* ewT[3] = {(__hip_bfloat16*)carve((size_t)1024 * 512 * 2),
                            (__hip_bfloat16*)carve((size_t)1024 * 512 * 2),
                            (__hip_bfloat16*)carve((size_t)1024 * 512 * 2)};
  __hip_bfloat16* dwT = (__hip_bfloat16*)carve((size_t)1024 * 256 * 2);

  // 1) precision conversion + weight transposes
  cvt_f32_bf16<<<4096, 256, 0, stream>>>(x, xb, B_ * T_ * F_);
  prep_conv_w<<<(256 * 320 + 255) / 256, 256, 0, stream>>>(cnn_w, cwT);
  for (int l = 0; l < 3; ++l)
    prep_enc_w<<<(1024 * 512) / 256, 256, 0, stream>>>(ew[l], eu[l], ewT[l]);
  prep_dec_w<<<(1024 * 256) / 256, 256, 0, stream>>>(cw, cu, dwT);

  // 2) conv1d + relu  (131072 x 256 output, 16384 blocks x 8 waves)
  conv_relu_wmma<<<16384, 256, 0, stream>>>(xb, cwT, cnn_b, y0);

  // 3) stacked encoder LSTMs, one fused WMMA kernel per timestep
  const __hip_bfloat16* yin = y0;
  __hip_bfloat16* yo_buf = y1;
  int p = 0;
  for (int l = 0; l < 3; ++l) {
    hipMemsetAsync(hpp[0], 0, (size_t)B_ * U_ * 2, stream);
    hipMemsetAsync(cpp[0], 0, (size_t)B_ * U_ * 4, stream);
    p = 0;
    for (int t = 0; t < T_; ++t) {
      __hip_bfloat16* yo = (l < 2) ? (yo_buf + (size_t)t * U_) : nullptr;
      lstm_step_wmma<<<32, 256, 0, stream>>>(
          yin + (size_t)t * U_, (size_t)T_ * U_,
          hpp[p], cpp[p], ewT[l], eb[l],
          hpp[1 - p], cpp[1 - p],
          yo, (size_t)T_ * U_, nullptr);
      p ^= 1;
    }
    if (l == 0) { yin = y1; yo_buf = y0; }
    else if (l == 1) { yin = y0; }
  }

  // 4) autoregressive decoder: h @ (cell_w + cell_u) folded into one weight
  for (int s = 0; s < S_; ++s) {
    lstm_step_wmma<<<32, 256, 0, stream>>>(
        nullptr, 0,
        hpp[p], cpp[p], dwT, cb,
        hpp[1 - p], cpp[1 - p],
        nullptr, 0, dec + (size_t)s * B_ * U_);
    p ^= 1;
  }

  // 5) dense head
  dense_out_kernel<<<(B_ * S_ * O_ + 255) / 256, 256, 0, stream>>>(dec, dw, db, out);
}
